// SlowFastSSMM_23184233464549
// MI455X (gfx1250) — matrix operand order repaired and checked by reading, not compiled, not run on hardware
//
#include <hip/hip_runtime.h>
#include <hip/hip_bf16.h>
#include <hip/hip_fp16.h>
#include <math.h>

typedef __attribute__((ext_vector_type(16))) _Float16 v16h;
typedef __attribute__((ext_vector_type(8)))  _Float16 v8h;
typedef __attribute__((ext_vector_type(8)))  float    v8f;

#define B_SZ     4
#define T_LEN    32768
#define L_SEQ    32
#define DELTA_S  16
#define NUM_SLOW 2047              // (T - L_S)/DELTA_S + 1
#define NSEQ     (B_SZ * NUM_SLOW) // 8188
#define GRU_H    64
#define G3       192
#define N_GRU    4
#define HIDDEN   32
#define SEQ_TILE 16
#define NTILES   12                // G3 / 16
#define KCH      2                 // GRU_H / 32
#define CHUNK    256
#define NCH      (T_LEN / CHUNK)   // 128

// Fast transcendentals: v_exp_f32 + v_rcp_f32 (no precise-division fixup chain)
__device__ __forceinline__ float fast_sigmoid(float x) {
    return __builtin_amdgcn_rcpf(1.0f + __expf(-x));
}
__device__ __forceinline__ float fast_tanh(float x) {
    // tanh(x) = 1 - 2/(e^{2x}+1)
    return 1.0f - 2.0f * __builtin_amdgcn_rcpf(__expf(2.0f * x) + 1.0f);
}

#define WMMA_F16(A, B, C) \
    __builtin_amdgcn_wmma_f32_16x16x32_f16(false, (A), false, (B), (short)0, (C), false, false)

// ---------------------------------------------------------------------------
// Kernel 1: slow path. One workgroup = 16 sequences, 4 waves.
// Wave wv owns gate tiles: r = tile wv, z = tile wv+4, n = tile wv+8.
// All waves run the identical, branch-free 12-WMMA sequence per timestep;
// weight (B) fragments live in registers for the whole layer.
// ---------------------------------------------------------------------------
__global__ __launch_bounds__(128) void gru_slow_kernel(
    const float* __restrict__ x,       const float* __restrict__ fc_in_w,
    const float* __restrict__ fc_in_b, const float* __restrict__ gru_wih,
    const float* __restrict__ gru_whh, const float* __restrict__ gru_bih,
    const float* __restrict__ gru_bhh, const float* __restrict__ out_fc_w,
    const float* __restrict__ out_fc_b,
    float* __restrict__ wsA, float* __restrict__ wsG)
{
    // Layer input/output sequence buffer (updated in place layer by layer)
    __shared__ __align__(32) _Float16 Xbuf[SEQ_TILE][L_SEQ][GRU_H];      // 64 KB
    // Weights, fragment-swizzled for WMMA B operand (32 contiguous bytes/lane)
    __shared__ __align__(32) _Float16 WihF[NTILES][KCH][32][16];         // 24 KB
    __shared__ __align__(32) _Float16 WhhF[NTILES][KCH][32][16];         // 24 KB
    // Pre-activation staging: tiles 0-3 r(sum), 4-7 z(sum), 8-11 inew, 12-15 hnew
    __shared__ float pre[16][SEQ_TILE][16];                              // 16 KB
    __shared__ __align__(32) _Float16 Hcur[SEQ_TILE][GRU_H];             // 2 KB
    __shared__ float br[GRU_H], bz[GRU_H], bin_[GRU_H], bhn[GRU_H];      // 1 KB

    const int tid  = threadIdx.x;
    const int lane = tid & 31;
    const int wv   = tid >> 5;
    const int wg   = blockIdx.x;

    // ---- layer-0 input: relu(x * fc_in_w + fc_in_b) ----
    for (int idx = tid; idx < SEQ_TILE * L_SEQ * GRU_H; idx += 128) {
        int s = idx / (L_SEQ * GRU_H);
        int t = (idx / GRU_H) % L_SEQ;
        int k = idx % GRU_H;
        int gs = wg * SEQ_TILE + s;
        float xv = 0.0f;
        if (gs < NSEQ) {
            int b = gs / NUM_SLOW, f = gs % NUM_SLOW;
            xv = x[b * T_LEN + f * DELTA_S + t];
        }
        float v = xv * fc_in_w[k] + fc_in_b[k];
        Xbuf[s][t][k] = (_Float16)(v > 0.0f ? v : 0.0f);
    }

    // Per-wave A-fragment geometry (16-bit A layout: two 8-half segments)
    const int arow  = lane & 15;
    const int khalf = (lane >> 4) << 3;         // 0 or 8
    const int mrow  = (lane >> 4) << 3;         // C/D layout: M = v + 8*(lane>=16)
    const int ncol  = lane & 15;

    for (int l = 0; l < N_GRU; ++l) {
        __syncthreads();  // previous layer compute done before LDS weight overwrite

        // ---- stage f32 weights -> f16 WMMA fragments in LDS ----
        const float* Wi = gru_wih + l * G3 * GRU_H;
        const float* Wh = gru_whh + l * G3 * GRU_H;
        for (int idx = tid; idx < NTILES * KCH * 32 * 16; idx += 128) {
            int e  = idx & 15;
            int ln = (idx >> 4) & 31;
            int kc = (idx >> 9) & (KCH - 1);
            int nt = idx >> 10;
            int g  = nt * 16 + (ln & 15);
            int k  = kc * 32 + ((e >> 3) << 4) + ((ln >> 4) << 3) + (e & 7);   // lanes>=16 carry K=8..15,24..31
            WihF[nt][kc][ln][e] = (_Float16)Wi[g * GRU_H + k];
            WhhF[nt][kc][ln][e] = (_Float16)Wh[g * GRU_H + k];
        }
        const float* bi = gru_bih + l * G3;
        const float* bh = gru_bhh + l * G3;
        for (int j = tid; j < GRU_H; j += 128) {
            br[j]   = bi[j] + bh[j];
            bz[j]   = bi[GRU_H + j] + bh[GRU_H + j];
            bin_[j] = bi[2 * GRU_H + j];
            bhn[j]  = bh[2 * GRU_H + j];
        }
        for (int idx = tid; idx < SEQ_TILE * GRU_H; idx += 128)
            ((_Float16*)Hcur)[idx] = (_Float16)0.0f;   // h0 = 0 each layer
        __syncthreads();

        // ---- hoist this wave's 12 B fragments into registers for the layer ----
        const v16h bRi0 = *(const v16h*)&WihF[wv    ][0][lane][0];
        const v16h bRi1 = *(const v16h*)&WihF[wv    ][1][lane][0];
        const v16h bRh0 = *(const v16h*)&WhhF[wv    ][0][lane][0];
        const v16h bRh1 = *(const v16h*)&WhhF[wv    ][1][lane][0];
        const v16h bZi0 = *(const v16h*)&WihF[wv + 4][0][lane][0];
        const v16h bZi1 = *(const v16h*)&WihF[wv + 4][1][lane][0];
        const v16h bZh0 = *(const v16h*)&WhhF[wv + 4][0][lane][0];
        const v16h bZh1 = *(const v16h*)&WhhF[wv + 4][1][lane][0];
        const v16h bNi0 = *(const v16h*)&WihF[wv + 8][0][lane][0];
        const v16h bNi1 = *(const v16h*)&WihF[wv + 8][1][lane][0];
        const v16h bNh0 = *(const v16h*)&WhhF[wv + 8][0][lane][0];
        const v16h bNh1 = *(const v16h*)&WhhF[wv + 8][1][lane][0];

        for (int t = 0; t < L_SEQ; ++t) {
            // ---- A fragments for x_t and h (branch-free, EXEC all ones) ----
            union { v16h v; v8h h[2]; } ax0, ax1, ah0, ah1;
            ax0.h[0] = *(const v8h*)&Xbuf[arow][t][ 0 + khalf];
            ax0.h[1] = *(const v8h*)&Xbuf[arow][t][16 + khalf];
            ax1.h[0] = *(const v8h*)&Xbuf[arow][t][32 + khalf];
            ax1.h[1] = *(const v8h*)&Xbuf[arow][t][48 + khalf];
            ah0.h[0] = *(const v8h*)&Hcur[arow][ 0 + khalf];
            ah0.h[1] = *(const v8h*)&Hcur[arow][16 + khalf];
            ah1.h[0] = *(const v8h*)&Hcur[arow][32 + khalf];
            ah1.h[1] = *(const v8h*)&Hcur[arow][48 + khalf];

            // r gate: x-path + h-path share one accumulator
            v8f accR = {};
            accR = WMMA_F16(ax0.v, bRi0, accR);
            accR = WMMA_F16(ax1.v, bRi1, accR);
            accR = WMMA_F16(ah0.v, bRh0, accR);
            accR = WMMA_F16(ah1.v, bRh1, accR);
            // z gate
            v8f accZ = {};
            accZ = WMMA_F16(ax0.v, bZi0, accZ);
            accZ = WMMA_F16(ax1.v, bZi1, accZ);
            accZ = WMMA_F16(ah0.v, bZh0, accZ);
            accZ = WMMA_F16(ah1.v, bZh1, accZ);
            // n gate: inew / hnew kept separate (n = tanh(inew + r*hnew))
            v8f accNI = {}, accNH = {};
            accNI = WMMA_F16(ax0.v, bNi0, accNI);
            accNI = WMMA_F16(ax1.v, bNi1, accNI);
            accNH = WMMA_F16(ah0.v, bNh0, accNH);
            accNH = WMMA_F16(ah1.v, bNh1, accNH);

            #pragma unroll
            for (int vi = 0; vi < 8; ++vi) {
                pre[wv     ][vi + mrow][ncol] = accR[vi];
                pre[wv + 4 ][vi + mrow][ncol] = accZ[vi];
                pre[wv + 8 ][vi + mrow][ncol] = accNI[vi];
                pre[wv + 12][vi + mrow][ncol] = accNH[vi];
            }
            __syncthreads();

            // ---- gate combine + state update (1024 elems over 128 threads) ----
            for (int idx = tid; idx < SEQ_TILE * GRU_H; idx += 128) {
                int s  = idx >> 6;
                int j  = idx & 63;
                int nt = j >> 4, c = j & 15;
                float r  = fast_sigmoid(pre[nt     ][s][c] + br[j]);
                float z  = fast_sigmoid(pre[4 + nt ][s][c] + bz[j]);
                float nn = fast_tanh(pre[8 + nt][s][c] + bin_[j] +
                                     r * (pre[12 + nt][s][c] + bhn[j]));
                float hold = (float)Hcur[s][j];
                float hv = (1.0f - z) * nn + z * hold;
                _Float16 hh = (_Float16)hv;
                Hcur[s][j]    = hh;
                Xbuf[s][t][j] = hh;   // in-place: next layer's input at time t
            }
            __syncthreads();
        }
    }

    // ---- epilogue: eps = h_last @ out_fc_w^T + b ; A = sigmoid(eps[:32]), g = eps[32:] ----
    for (int idx = tid; idx < SEQ_TILE * 2 * HIDDEN; idx += 128) {
        int s = idx >> 6;
        int o = idx & 63;
        float acc = out_fc_b[o];
        const float* wrow = out_fc_w + o * GRU_H;
        #pragma unroll 8
        for (int k = 0; k < GRU_H; ++k)
            acc += (float)Hcur[s][k] * wrow[k];
        int gs = wg * SEQ_TILE + s;
        if (gs < NSEQ) {
            int b = gs / NUM_SLOW, f = gs % NUM_SLOW;
            if (o < HIDDEN) wsA[(b * NUM_SLOW + f) * HIDDEN + o]            = fast_sigmoid(acc);
            else            wsG[(b * NUM_SLOW + f) * HIDDEN + (o - HIDDEN)] = acc;
        }
    }
}

// ---------------------------------------------------------------------------
// Fast path: chunked parallel scan of h_t = A_t*h + u_t (32 lanes = 32 hidden)
// ---------------------------------------------------------------------------
__global__ __launch_bounds__(32) void chunk_local_kernel(
    const float* __restrict__ x, const float* __restrict__ fin_w, const float* __restrict__ fin_b,
    const float* __restrict__ wsA, const float* __restrict__ wsG,
    float* __restrict__ wsAp, float* __restrict__ wsU)
{
    int b = blockIdx.x / NCH;
    int c = blockIdx.x % NCH;
    int h = threadIdx.x;
    float fw = fin_w[h], fb = fin_b[h];
    float ap = 1.0f, u = 0.0f;
    int t0 = c * CHUNK;
    for (int blk = 0; blk < CHUNK / DELTA_S; ++blk) {
        int tb  = t0 / DELTA_S + blk;
        int idx = tb - 1; if (idx < 0) idx = 0; if (idx > NUM_SLOW - 1) idx = NUM_SLOW - 1;
        float A = wsA[(b * NUM_SLOW + idx) * HIDDEN + h];
        float g = wsG[(b * NUM_SLOW + idx) * HIDDEN + h];
        const float* xp = x + b * T_LEN + t0 + blk * DELTA_S;
        #pragma unroll
        for (int j = 0; j < DELTA_S; ++j) {
            float uu = (xp[j] * fw + fb) * g;
            u  = A * u + uu;
            ap *= A;
        }
    }
    wsAp[(b * NCH + c) * HIDDEN + h] = ap;
    wsU [(b * NCH + c) * HIDDEN + h] = u;
}

__global__ __launch_bounds__(32) void chunk_scan_kernel(
    const float* __restrict__ wsAp, const float* __restrict__ wsU, float* __restrict__ wsS)
{
    int b = blockIdx.x;
    int h = threadIdx.x;
    float hs = 0.0f;
    for (int c = 0; c < NCH; ++c) {
        wsS[(b * NCH + c) * HIDDEN + h] = hs;
        hs = wsAp[(b * NCH + c) * HIDDEN + h] * hs + wsU[(b * NCH + c) * HIDDEN + h];
    }
}

__global__ __launch_bounds__(32) void apply_kernel(
    const float* __restrict__ x, const float* __restrict__ fin_w, const float* __restrict__ fin_b,
    const float* __restrict__ fout_w, const float* __restrict__ fout_b,
    const float* __restrict__ wsA, const float* __restrict__ wsG, const float* __restrict__ wsS,
    float* __restrict__ out)
{
    int b = blockIdx.x / NCH;
    int c = blockIdx.x % NCH;
    int h = threadIdx.x;
    float fw = fin_w[h], fb = fin_b[h], ow = fout_w[h], ob = fout_b[0];
    float hs = wsS[(b * NCH + c) * HIDDEN + h];
    int t0 = c * CHUNK;
    for (int blk = 0; blk < CHUNK / DELTA_S; ++blk) {
        int tb  = t0 / DELTA_S + blk;
        int idx = tb - 1; if (idx < 0) idx = 0; if (idx > NUM_SLOW - 1) idx = NUM_SLOW - 1;
        float A = wsA[(b * NUM_SLOW + idx) * HIDDEN + h];
        float g = wsG[(b * NUM_SLOW + idx) * HIDDEN + h];
        const float* xp = x + b * T_LEN + t0 + blk * DELTA_S;
        float* op = out + b * T_LEN + t0 + blk * DELTA_S;
        #pragma unroll
        for (int j = 0; j < DELTA_S; ++j) {
            float uu = (xp[j] * fw + fb) * g;
            hs = A * hs + uu;
            float v = hs * ow;
            #pragma unroll
            for (int m = 16; m >= 1; m >>= 1) v += __shfl_xor(v, m, 32);
            if (h == 0) op[j] = v + ob;
        }
    }
}

extern "C" void kernel_launch(void* const* d_in, const int* in_sizes, int n_in,
                              void* d_out, int out_size, void* d_ws, size_t ws_size,
                              hipStream_t stream) {
    (void)in_sizes; (void)n_in; (void)out_size; (void)ws_size;
    const float* x        = (const float*)d_in[0];
    const float* fc_in_w  = (const float*)d_in[1];
    const float* fc_in_b  = (const float*)d_in[2];
    const float* gru_wih  = (const float*)d_in[3];
    const float* gru_whh  = (const float*)d_in[4];
    const float* gru_bih  = (const float*)d_in[5];
    const float* gru_bhh  = (const float*)d_in[6];
    const float* out_fc_w = (const float*)d_in[7];
    const float* out_fc_b = (const float*)d_in[8];
    const float* fin_w    = (const float*)d_in[9];
    const float* fin_b    = (const float*)d_in[10];
    const float* fout_w   = (const float*)d_in[11];
    const float* fout_b   = (const float*)d_in[12];
    float* out = (float*)d_out;

    float* wsA  = (float*)d_ws;
    float* wsG  = wsA  + (size_t)B_SZ * NUM_SLOW * HIDDEN;
    float* wsAp = wsG  + (size_t)B_SZ * NUM_SLOW * HIDDEN;
    float* wsU  = wsAp + (size_t)B_SZ * NCH * HIDDEN;
    float* wsS  = wsU  + (size_t)B_SZ * NCH * HIDDEN;

    int nwg = (NSEQ + SEQ_TILE - 1) / SEQ_TILE;  // 512 workgroups of 4 waves
    gru_slow_kernel<<<nwg, 128, 0, stream>>>(x, fc_in_w, fc_in_b, gru_wih, gru_whh,
                                             gru_bih, gru_bhh, out_fc_w, out_fc_b, wsA, wsG);
    chunk_local_kernel<<<B_SZ * NCH, 32, 0, stream>>>(x, fin_w, fin_b, wsA, wsG, wsAp, wsU);
    chunk_scan_kernel<<<B_SZ, 32, 0, stream>>>(wsAp, wsU, wsS);
    apply_kernel<<<B_SZ * NCH, 32, 0, stream>>>(x, fin_w, fin_b, fout_w, fout_b,
                                                wsA, wsG, wsS, out);
}
